// Attention_80204219285917
// MI455X (gfx1250) — compile-verified, matrix operand order checked by reading, not run on hardware
//
#include <hip/hip_runtime.h>
#include <hip/hip_bf16.h>

// ---------------------------------------------------------------------------
// Multi-head attention for MI455X (gfx1250, wave32, WMMA f32_16x16x32_f16).
// Pipeline: cast->f16 | Q/K/V proj GEMMs (WMMA, double-buffered LDS) |
// flash attention (WMMA, online softmax) | output proj GEMM (WMMA, fp32 out).
// ~69 GFLOP total, ~120MB minimum HBM traffic -> roughly balanced with f16
// matrix cores; scalar fp32 would be hopelessly compute-bound.
// ---------------------------------------------------------------------------

typedef __attribute__((ext_vector_type(16))) _Float16 v16h;
typedef __attribute__((ext_vector_type(8)))  float    v8f;

#define WMMA16(a, b, c) \
  __builtin_amdgcn_wmma_f32_16x16x32_f16(false, (a), false, (b), (short)0, (c), false, false)

constexpr int BB = 2;      // batch
constexpr int S  = 2048;   // seq len
constexpr int D  = 1024;   // model dim
constexpr int H  = 16;     // heads
constexpr int HD = 64;     // head dim
constexpr int MT = BB * S; // 4096 tokens

// Load one v16h WMMA operand (16-bit A/B layout): caller passes base already
// offset by kA = (lane<16 ? 0 : 8); halves 0..7 = base[0..7], 8..15 = base[16..23].
__device__ __forceinline__ v16h load_op16(const _Float16* p) {
  union { v16h v; float4 f4[2]; } u;
  u.f4[0] = *(const float4*)(p);
  u.f4[1] = *(const float4*)(p + 16);
  return u.v;
}

// ---------------------------------------------------------------------------
// f32 -> f16 cast, 4 elements per thread
// ---------------------------------------------------------------------------
__global__ void cast_f32_to_f16(const float* __restrict__ src,
                                _Float16* __restrict__ dst, int n4) {
  int i = blockIdx.x * blockDim.x + threadIdx.x;
  if (i < n4) {
    float4 v = ((const float4*)src)[i];
    union { _Float16 h[4]; unsigned long long u; } w;
    w.h[0] = (_Float16)v.x; w.h[1] = (_Float16)v.y;
    w.h[2] = (_Float16)v.z; w.h[3] = (_Float16)v.w;
    ((unsigned long long*)dst)[i] = w.u;
  }
}

// ---------------------------------------------------------------------------
// GEMM: C[M=4096][N=1024] = A[4096][1024](f16) x B[1024][1024](f16) + bias(f32)
// MODE 0: store f16 row-major      (Q, K projections)
// MODE 1: store f16 as Vt[b,h,d,s] (V projection, pre-transposed for P@V)
// MODE 2: store f32 row-major      (final output projection)
// Block tile 128x128, 8 waves (4 along M x 2 along N), K-step 32,
// double-buffered LDS: one barrier per K-step, global loads for step k+1
// staged in registers while WMMAs consume buffer k.
// ---------------------------------------------------------------------------
template <int MODE>
__global__ __launch_bounds__(256) void gemm_f16_wmma(
    const _Float16* __restrict__ A, const _Float16* __restrict__ Bw,
    const float* __restrict__ bias, void* __restrict__ out) {
  constexpr int K = 1024, N = 1024;
  __shared__ _Float16 lds_a[2][128 * 32];  // [m][k]
  __shared__ _Float16 lds_b[2][128 * 32];  // [n][k] (transposed on the way in)

  const int tid  = threadIdx.x;
  const int wave = tid >> 5, lane = tid & 31;
  const int l15  = lane & 15, hi = lane >> 4;
  const int kA   = hi ? 8 : 0;
  const int wm   = wave & 3;   // 4 wave-rows of 32
  const int wn   = wave >> 2;  // 2 wave-cols of 64
  const int m0   = blockIdx.y * 128, n0 = blockIdx.x * 128;

  // per-thread staging assignment
  const int arow = tid >> 1, akc = (tid & 1) * 16;  // A: 128 rows x 32
  const int bkr = tid >> 3, bch = tid & 7;          // B: 32 rows x 8 col-chunks

  v8f acc[2][4];
  const v8f vzero = {0.f, 0.f, 0.f, 0.f, 0.f, 0.f, 0.f, 0.f};
#pragma unroll
  for (int i = 0; i < 2; i++)
#pragma unroll
    for (int j = 0; j < 4; j++) acc[i][j] = vzero;

  float4 ra0, ra1, rb0, rb1;
  auto g_load = [&](int k0) {
    const float4* asrc = (const float4*)(A + (size_t)(m0 + arow) * K + k0 + akc);
    ra0 = asrc[0];
    ra1 = asrc[1];
    const _Float16* bsrc = Bw + (size_t)(k0 + bkr) * N + n0 + bch * 16;
    rb0 = *(const float4*)(bsrc);
    rb1 = *(const float4*)(bsrc + 8);
  };
  auto l_store = [&](int buf) {
    *(float4*)(&lds_a[buf][arow * 32 + akc])     = ra0;
    *(float4*)(&lds_a[buf][arow * 32 + akc + 8]) = ra1;
    union { float4 f4[2]; _Float16 h[16]; } u;
    u.f4[0] = rb0;
    u.f4[1] = rb1;
#pragma unroll
    for (int i = 0; i < 16; i++) lds_b[buf][(bch * 16 + i) * 32 + bkr] = u.h[i];
  };

  g_load(0);
  l_store(0);
  __syncthreads();

  for (int k0 = 0; k0 < K; k0 += 32) {
    const int cur = (k0 >> 5) & 1, nxt = cur ^ 1;
    const bool more = (k0 + 32) < K;
    if (more) g_load(k0 + 32);  // overlap next global fetch with WMMA

    v16h aop[2], bop[4];
#pragma unroll
    for (int mt = 0; mt < 2; mt++)
      aop[mt] = load_op16(&lds_a[cur][(wm * 32 + mt * 16 + l15) * 32 + kA]);
#pragma unroll
    for (int nt = 0; nt < 4; nt++)
      bop[nt] = load_op16(&lds_b[cur][(wn * 64 + nt * 16 + l15) * 32 + kA]);
#pragma unroll
    for (int mt = 0; mt < 2; mt++)
#pragma unroll
      for (int nt = 0; nt < 4; nt++)
        acc[mt][nt] = WMMA16(aop[mt], bop[nt], acc[mt][nt]);

    if (more) l_store(nxt);  // other buffer: no conflict with current readers
    __syncthreads();
  }

  // Epilogue: add bias, store per mode. C layout: VGPR r holds row r + 8*hi,
  // lane%16 holds column within the 16x16 tile.
#pragma unroll
  for (int mt = 0; mt < 2; mt++) {
#pragma unroll
    for (int nt = 0; nt < 4; nt++) {
      int col = n0 + wn * 64 + nt * 16 + l15;
      float bv = bias[col];
      union { v8f v; float f[8]; } u;
      u.v = acc[mt][nt];
#pragma unroll
      for (int r = 0; r < 8; r++) {
        int row = m0 + wm * 32 + mt * 16 + r + hi * 8;
        float val = u.f[r] + bv;
        if (MODE == 0) {
          ((_Float16*)out)[(size_t)row * N + col] = (_Float16)val;
        } else if (MODE == 1) {
          int hh = col >> 6, dd = col & 63;        // head, dim-in-head
          int b = row >> 11, s = row & (S - 1);    // batch, position
          ((_Float16*)out)[((size_t)(b * H + hh) * HD + dd) * S + s] = (_Float16)val;
        } else {
          ((float*)out)[(size_t)row * N + col] = val;
        }
      }
    }
  }
}

// ---------------------------------------------------------------------------
// Flash attention: one wave = 16 query rows of one (b,h); block = 8 waves
// = 128 query rows. Online softmax over 32-key chunks. No cross-wave sync.
// K/V total 16MB f16 -> resident in 192MB L2, so no LDS staging for K/V.
// ---------------------------------------------------------------------------
__global__ __launch_bounds__(256) void flash_attn_wmma(
    const _Float16* __restrict__ Qp, const _Float16* __restrict__ Kp,
    const _Float16* __restrict__ Vt, const int* __restrict__ mask,
    _Float16* __restrict__ ctx) {
  __shared__ _Float16 lds_p[8][16 * 32];  // wave-private P staging (fp32->f16 relayout)

  const int tid  = threadIdx.x;
  const int wave = tid >> 5, lane = tid & 31;
  const int l15  = lane & 15, hi = lane >> 4;
  const int kA   = hi ? 8 : 0;

  const int qb = blockIdx.x & 15;  // S/128 = 16 query blocks
  const int bh = blockIdx.x >> 4;  // b*H + h
  const int b  = bh >> 4, h = bh & 15;
  const int sq = qb * 128 + wave * 16;  // first query row (within sequence)

  // Q rows held in registers: A-operand layout, d split into two K=32 chunks
  const _Float16* qbase = Qp + ((size_t)(b * S + sq + l15)) * D + h * HD;
  v16h aq0 = load_op16(qbase + kA);
  v16h aq1 = load_op16(qbase + 32 + kA);

  const v8f vzero = {0.f, 0.f, 0.f, 0.f, 0.f, 0.f, 0.f, 0.f};
  union { v8f v; float f[8]; } acc[4];
  float mrow[8], lrow[8];
#pragma unroll
  for (int t = 0; t < 4; t++) acc[t].v = vzero;
#pragma unroll
  for (int r = 0; r < 8; r++) { mrow[r] = -INFINITY; lrow[r] = 0.f; }

  _Float16* myp = &lds_p[wave][0];

  for (int kc = 0; kc < S; kc += 32) {
    // ---- scores: two 16x16 tiles (32 keys), contraction over HD=64 = 2xK32
    float p[2][8];
#pragma unroll
    for (int j = 0; j < 2; j++) {
      int key = kc + j * 16 + l15;
      const _Float16* kb = Kp + ((size_t)(b * S + key)) * D + h * HD;
      v16h bk0 = load_op16(kb + kA);
      v16h bk1 = load_op16(kb + 32 + kA);
      v8f sc = vzero;
      sc = WMMA16(aq0, bk0, sc);
      sc = WMMA16(aq1, bk1, sc);
      union { v8f v; float f[8]; } us;
      us.v = sc;
#pragma unroll
      for (int r = 0; r < 8; r++) {
        int q = sq + r + hi * 8;
        int mv = mask[((size_t)b * S + q) * S + (kc + j * 16 + l15)];
        float val = us.f[r] * 0.125f;  // 1/sqrt(64)
        p[j][r] = mv ? val : -1.0e9f;
      }
    }
    // ---- online softmax; a row spans the 16 lanes of one half (xor<16 stays in half)
#pragma unroll
    for (int r = 0; r < 8; r++) {
      float m = fmaxf(p[0][r], p[1][r]);
      m = fmaxf(m, __shfl_xor(m, 1, 32));
      m = fmaxf(m, __shfl_xor(m, 2, 32));
      m = fmaxf(m, __shfl_xor(m, 4, 32));
      m = fmaxf(m, __shfl_xor(m, 8, 32));
      float mnew = fmaxf(mrow[r], m);
      float corr = __expf(mrow[r] - mnew);
      mrow[r] = mnew;
      float p0 = __expf(p[0][r] - mnew);
      float p1 = __expf(p[1][r] - mnew);
      float sum = p0 + p1;
      sum += __shfl_xor(sum, 1, 32);
      sum += __shfl_xor(sum, 2, 32);
      sum += __shfl_xor(sum, 4, 32);
      sum += __shfl_xor(sum, 8, 32);
      lrow[r] = lrow[r] * corr + sum;
#pragma unroll
      for (int t = 0; t < 4; t++) acc[t].f[r] *= corr;
      // C-layout -> row-major f16 P tile in wave-private LDS
      myp[(r + hi * 8) * 32 + l15]      = (_Float16)p0;
      myp[(r + hi * 8) * 32 + 16 + l15] = (_Float16)p1;
    }
    // LDS is in-order per wave; wait so the A-operand gather sees all 32 lanes' stores
    asm volatile("s_wait_dscnt 0" ::: "memory");
    v16h ap = load_op16(&myp[l15 * 32 + kA]);
    // ---- P(16x32) @ V(32x64): Vt rows are contiguous along keys
#pragma unroll
    for (int t = 0; t < 4; t++) {
      const _Float16* vb = Vt + ((size_t)bh * HD + t * 16 + l15) * S + kc;
      v16h bv = load_op16(vb + kA);
      acc[t].v = WMMA16(ap, bv, acc[t].v);
    }
    asm volatile("" ::: "memory");
  }

  // ---- normalize and scatter into context [token][h*64 + d]
#pragma unroll
  for (int t = 0; t < 4; t++) {
#pragma unroll
    for (int r = 0; r < 8; r++) {
      int q = sq + r + hi * 8;
      float o = acc[t].f[r] / lrow[r];
      ctx[((size_t)(b * S + q)) * D + h * HD + t * 16 + l15] = (_Float16)o;
    }
  }
}

// ---------------------------------------------------------------------------
// Host-side launch
// ---------------------------------------------------------------------------
extern "C" void kernel_launch(void* const* d_in, const int* in_sizes, int n_in,
                              void* d_out, int out_size, void* d_ws, size_t ws_size,
                              hipStream_t stream) {
  const float* q_in  = (const float*)d_in[0];
  const float* k_in  = (const float*)d_in[1];
  const float* v_in  = (const float*)d_in[2];
  const int*   mask  = (const int*)d_in[3];
  const float* wq    = (const float*)d_in[4];
  const float* wqb   = (const float*)d_in[5];
  const float* wk    = (const float*)d_in[6];
  const float* wkb   = (const float*)d_in[7];
  const float* wv    = (const float*)d_in[8];
  const float* wvb   = (const float*)d_in[9];
  const float* wo    = (const float*)d_in[10];
  const float* wob   = (const float*)d_in[11];

  // Workspace carving (64 MB total, 256B aligned)
  char* ws = (char*)d_ws;
  size_t off = 0;
  auto alloc = [&](size_t bytes) {
    void* p = ws + off;
    off += (bytes + 255) & ~(size_t)255;
    return p;
  };
  const size_t IN_E = (size_t)MT * D;   // 4194304
  const size_t W_E  = (size_t)D * D;    // 1048576
  _Float16* qh  = (_Float16*)alloc(IN_E * 2);
  _Float16* kh  = (_Float16*)alloc(IN_E * 2);
  _Float16* vh  = (_Float16*)alloc(IN_E * 2);
  _Float16* wqh = (_Float16*)alloc(W_E * 2);
  _Float16* wkh = (_Float16*)alloc(W_E * 2);
  _Float16* wvh = (_Float16*)alloc(W_E * 2);
  _Float16* woh = (_Float16*)alloc(W_E * 2);
  _Float16* Qp  = (_Float16*)alloc(IN_E * 2);
  _Float16* Kp  = (_Float16*)alloc(IN_E * 2);
  _Float16* Vt  = (_Float16*)alloc(IN_E * 2);   // [b][h][d][s]
  _Float16* ctx = (_Float16*)alloc(IN_E * 2);

  // 1) casts
  int n4i = (int)(IN_E / 4), n4w = (int)(W_E / 4);
  cast_f32_to_f16<<<(n4i + 255) / 256, 256, 0, stream>>>(q_in, qh, n4i);
  cast_f32_to_f16<<<(n4i + 255) / 256, 256, 0, stream>>>(k_in, kh, n4i);
  cast_f32_to_f16<<<(n4i + 255) / 256, 256, 0, stream>>>(v_in, vh, n4i);
  cast_f32_to_f16<<<(n4w + 255) / 256, 256, 0, stream>>>(wq, wqh, n4w);
  cast_f32_to_f16<<<(n4w + 255) / 256, 256, 0, stream>>>(wk, wkh, n4w);
  cast_f32_to_f16<<<(n4w + 255) / 256, 256, 0, stream>>>(wv, wvh, n4w);
  cast_f32_to_f16<<<(n4w + 255) / 256, 256, 0, stream>>>(wo, woh, n4w);

  // 2) projections
  dim3 gg(D / 128, MT / 128);  // (8, 32)
  gemm_f16_wmma<0><<<gg, 256, 0, stream>>>(qh, wqh, wqb, (void*)Qp);
  gemm_f16_wmma<0><<<gg, 256, 0, stream>>>(kh, wkh, wkb, (void*)Kp);
  gemm_f16_wmma<1><<<gg, 256, 0, stream>>>(vh, wvh, wvb, (void*)Vt);

  // 3) attention: 512 blocks = B*H * (S/128)
  flash_attn_wmma<<<BB * H * (S / 128), 256, 0, stream>>>(Qp, Kp, Vt, mask, ctx);

  // 4) output projection -> fp32 d_out
  gemm_f16_wmma<2><<<gg, 256, 0, stream>>>(ctx, woh, wob, d_out);
}
